// DWT_Attention_17042430230760
// MI455X (gfx1250) — matrix operand order repaired and checked by reading, not compiled, not run on hardware
//
#include <hip/hip_runtime.h>
#include <hip/hip_bf16.h>

// ---------------------------------------------------------------------------
// DWT-Haar attention block for MI455X (gfx1250, wave32, WMMA).
// Heavy convs (grouped 3x3, all 1x1) -> implicit GEMM with
// v_wmma_f32_16x16x32_bf16; fragments staged in LDS in fragment element order
// so each lane's 16-bf16 fragment is a 32B contiguous load (2x ds_load_b128).
// All staging is pair-packed (bf16x2 -> ds_store_b32, float2 global reads).
// Tiny 8x8 attention, DWT/IDWT, depthwise -> bandwidth-bound VALU kernels.
// Workspace requirement: ~18 * (4*64*65536) floats + 10KB  ~= 1.21 GB.
// ---------------------------------------------------------------------------

typedef __attribute__((ext_vector_type(16))) __bf16 v16bf;
typedef __attribute__((ext_vector_type(2)))  __bf16 bf16x2;
typedef __attribute__((ext_vector_type(8)))  float  v8f;

#define BZ    4
#define DIMC  64
#define HEADS 8
#define HF    512
#define WF    512
#define HH_   256
#define WH_   256
#define NPIX  (HH_ * WH_)          // 65536

__device__ __forceinline__ __bf16 f2bf(float f) { return (__bf16)f; }

__device__ __forceinline__ v8f wmma_bf16(v16bf a, v16bf b, v8f c) {
    // (neg_a, A, neg_b, B, c_mod, C, reuse_a, reuse_b)
    return __builtin_amdgcn_wmma_f32_16x16x32_bf16(false, a, false, b,
                                                   (short)0, c, false, false);
}

// A-fragment K permutation: fragment slot t (= half*16 + j) -> K offset
//   K = half*8 + (j&7) + ((j&8)<<1)  ==  (t&7) | ((t&8)<<1) | ((t&16)>>1)
// Note: for even t, kperm(t+1) == kperm(t) + 1  (pairs stay adjacent).
__device__ __forceinline__ int kperm(int t) {
    return (t & 7) | ((t & 8) << 1) | ((t & 16) >> 1);
}

// ---------------------------------------------------------------------------
// Haar DWT: x (B,64,512,512) -> LL/LH/HL/HH (B,64,256,256). float2 loads.
// ---------------------------------------------------------------------------
__global__ void dwt_haar(const float* __restrict__ x, float* __restrict__ LL,
                         float* __restrict__ LH, float* __restrict__ HL,
                         float* __restrict__ HH) {
    size_t i = (size_t)blockIdx.x * 256 + threadIdx.x;   // over B*64*NPIX
    int xj = (int)(i % WH_);
    int yi = (int)((i / WH_) % HH_);
    size_t bc = i / NPIX;
    const float* xb = x + bc * (size_t)HF * WF;
    float2 e = ((const float2*)(xb + (size_t)(2 * yi) * WF))[xj];
    float2 o = ((const float2*)(xb + (size_t)(2 * yi + 1) * WF))[xj];
    float a = e.x, b = e.y, c = o.x, d = o.y;
    LL[i] = ( a + b + c + d) * 0.5f;
    LH[i] = (-a - b + c + d) * 0.5f;
    HL[i] = (-a + b - c + d) * 0.5f;
    HH[i] = ( a - b - c + d) * 0.5f;
}

// ---------------------------------------------------------------------------
// Grouped 3x3 conv (SAME), Cin=64/group, Cout=64/group, implicit GEMM.
// grid: x = WH_/64 strips, y = HH_ rows, z = b*G + g.  256 threads / 8 waves.
// Block tile: M=64 x N=64 pixels; each wave owns one 16x16 M-tile at two N
// positions (2 accumulators). K = 576 in steps of 32.
// ---------------------------------------------------------------------------
__global__ void __launch_bounds__(256)
conv3x3_wmma(const float* __restrict__ in0, const float* __restrict__ in1,
             const float* __restrict__ in2, const float* __restrict__ w,
             float* __restrict__ out, int G, int relu) {
    __shared__ __align__(32) __bf16 sP[64][3][68];   // input patch ci x kh x (64+2)
    __shared__ __align__(32) __bf16 sWf[64][32];     // frag-ordered weight K-tile
    __shared__ __align__(32) __bf16 sBf[64][32];     // frag-ordered im2col (n, kk)

    const int tid  = threadIdx.x;
    const int lane = tid & 31;
    const int wv   = tid >> 5;
    const int x0   = blockIdx.x * 64;
    const int y    = blockIdx.y;
    const int g    = blockIdx.z % G;
    const int b    = blockIdx.z / G;

    const float* in  = (g == 0) ? in0 : ((g == 1) ? in1 : in2);
    const float* inb = in + (size_t)b * 64 * NPIX;

    // Stage input patch (zero-padded borders), f32 -> bf16, pair-packed.
    for (int idx = tid; idx < 64 * 3 * 33; idx += 256) {
        int xp = (idx % 33) * 2;           // xx, xx+1 (covers 0..65)
        int r  = (idx / 33) % 3;
        int ci = idx / (33 * 3);
        int gy = y + r - 1;
        float v0 = 0.f, v1 = 0.f;
        if (gy >= 0 && gy < HH_) {
            const float* src = inb + (size_t)ci * NPIX + (size_t)gy * WH_;
            int gx0 = x0 + xp - 1;
            if (gx0 >= 0 && gx0 < WH_)         v0 = src[gx0];
            if (gx0 + 1 >= 0 && gx0 + 1 < WH_) v1 = src[gx0 + 1];
        }
        bf16x2 t; t[0] = f2bf(v0); t[1] = f2bf(v1);
        *(bf16x2*)&sP[ci][r][xp] = t;
    }
    __syncthreads();

    const int Mt   = (wv & 3) * 16;
    const int Nb   = (wv >> 2) * 16;     // tiles at Nb and Nb+32
    const int half = lane >> 4;
    const int l16  = lane & 15;

    v8f acc0 = {};
    v8f acc1 = {};
    const float* wg = w + (size_t)(g * 64) * 576;   // OIHW flat: K = ci*9+kh*3+kw

    for (int k0 = 0; k0 < 576; k0 += 32) {
        // Stage weights in fragment order (float2 read, bf16x2 packed store).
        for (int idx = tid; idx < 64 * 16; idx += 256) {
            int m  = idx >> 4;
            int tp = (idx & 15) * 2;
            float2 wv2 = *(const float2*)(wg + (size_t)m * 576 + k0 + tp);
            bf16x2 t; t[0] = f2bf(wv2.x); t[1] = f2bf(wv2.y);
            *(bf16x2*)&sWf[m][tp] = t;
        }
        // Stage im2col tile transposed: sBf[n][kk] = patch(pixel n, K=k0+kk).
        for (int idx = tid; idx < 64 * 16; idx += 256) {
            int n  = idx >> 4;
            int kp = (idx & 15) * 2;
            int k  = k0 + kp;
            int ci0 = k / 9,       r0 = k % 9;
            int ci1 = (k + 1) / 9, r1 = (k + 1) % 9;
            bf16x2 t;
            t[0] = sP[ci0][r0 / 3][n + (r0 % 3)];
            t[1] = sP[ci1][r1 / 3][n + (r1 % 3)];
            *(bf16x2*)&sBf[n][kp] = t;
        }
        __syncthreads();

        v16bf av  = *(const v16bf*)&sWf[Mt + l16][half * 16];
        v16bf bv0 = *(const v16bf*)&sBf[Nb + l16][half * 16];
        v16bf bv1 = *(const v16bf*)&sBf[Nb + 32 + l16][half * 16];
        acc0 = wmma_bf16(av, bv0, acc0);
        acc1 = wmma_bf16(av, bv1, acc1);
        __syncthreads();
    }

    // D: row = Mt + half*8 + r, cols = Nb + l16 and Nb + 32 + l16.
    float* ob = out + ((size_t)(b * G + g) * 64) * NPIX + (size_t)y * WH_ + x0;
#pragma unroll
    for (int r = 0; r < 8; ++r) {
        float v0 = acc0[r];
        float v1 = acc1[r];
        if (relu) { v0 = fmaxf(v0, 0.f); v1 = fmaxf(v1, 0.f); }
        float* row = ob + (size_t)(Mt + half * 8 + r) * NPIX;
        row[Nb + l16]      = v0;
        row[Nb + 32 + l16] = v1;
    }
}

// ---------------------------------------------------------------------------
// 1x1 conv as GEMM: out(b,co,n) = sum_ci W[co][ci] * in(b,ci,n).
// grid: x = NPIX/64, y = Cout/64, z = B. 256 threads / 8 waves, 2 N-tiles/wave.
// ---------------------------------------------------------------------------
__global__ void __launch_bounds__(256)
conv1x1_wmma(const float* __restrict__ in, const float* __restrict__ w,
             float* __restrict__ out, int Cin, int Cout, int relu) {
    __shared__ __align__(32) __bf16 sWf[64][32];     // frag-ordered weight K-tile
    __shared__ __align__(32) __bf16 sXf[64][144];    // X transposed: [n][ci], Cin<=128
                                                     // row stride 288B (32B multiple)
    const int tid  = threadIdx.x;
    const int lane = tid & 31;
    const int wv   = tid >> 5;
    const int n0   = blockIdx.x * 64;
    const int cb   = blockIdx.y * 64;
    const int b    = blockIdx.z;

    const float* inb = in + (size_t)b * Cin * NPIX;
    // Transposed staging, pair-packed along ci (both reads lane-coalesced).
    for (int idx = tid; idx < (Cin >> 1) * 64; idx += 256) {
        int nn = idx & 63;
        int ci = (idx >> 6) * 2;
        float a = inb[(size_t)ci * NPIX + n0 + nn];
        float c = inb[(size_t)(ci + 1) * NPIX + n0 + nn];
        bf16x2 t; t[0] = f2bf(a); t[1] = f2bf(c);
        *(bf16x2*)&sXf[nn][ci] = t;
    }
    __syncthreads();

    const int Mt   = (wv & 3) * 16;
    const int Nb   = (wv >> 2) * 16;
    const int half = lane >> 4;
    const int l16  = lane & 15;

    v8f acc0 = {};
    v8f acc1 = {};
    for (int k0 = 0; k0 < Cin; k0 += 32) {
        for (int idx = tid; idx < 64 * 16; idx += 256) {
            int m  = idx >> 4;
            int tp = (idx & 15) * 2;
            float2 wv2 = *(const float2*)(w + (size_t)(cb + m) * Cin + k0 + tp);
            bf16x2 t; t[0] = f2bf(wv2.x); t[1] = f2bf(wv2.y);
            *(bf16x2*)&sWf[m][tp] = t;
        }
        __syncthreads();

        v16bf av  = *(const v16bf*)&sWf[Mt + l16][half * 16];
        v16bf bv0 = *(const v16bf*)&sXf[Nb + l16][k0 + half * 16];
        v16bf bv1 = *(const v16bf*)&sXf[Nb + 32 + l16][k0 + half * 16];
        acc0 = wmma_bf16(av, bv0, acc0);
        acc1 = wmma_bf16(av, bv1, acc1);
        __syncthreads();
    }

    float* ob = out + ((size_t)b * Cout + cb) * NPIX + n0;
#pragma unroll
    for (int r = 0; r < 8; ++r) {
        float v0 = acc0[r];
        float v1 = acc1[r];
        if (relu) { v0 = fmaxf(v0, 0.f); v1 = fmaxf(v1, 0.f); }
        float* row = ob + (size_t)(Mt + half * 8 + r) * NPIX;
        row[Nb + l16]      = v0;
        row[Nb + 32 + l16] = v1;
    }
}

// ---------------------------------------------------------------------------
// Depthwise 3x3 (192 channels), SAME. Bandwidth-bound streaming kernel.
// ---------------------------------------------------------------------------
__global__ void dwconv3x3(const float* __restrict__ in, const float* __restrict__ w,
                          float* __restrict__ out) {
    size_t i = (size_t)blockIdx.x * 256 + threadIdx.x;   // over B*192*NPIX
    int x = (int)(i % WH_);
    int y = (int)((i / WH_) % HH_);
    int c = (int)((i / NPIX) % 192);
    const float* ib = in + (i / NPIX) * (size_t)NPIX;
    const float* wc = w + c * 9;
    float s = 0.f;
#pragma unroll
    for (int kh = 0; kh < 3; ++kh) {
        int yy = y + kh - 1;
        if (yy < 0 || yy >= HH_) continue;
#pragma unroll
        for (int kw = 0; kw < 3; ++kw) {
            int xx = x + kw - 1;
            if (xx < 0 || xx >= WH_) continue;
            s += wc[kh * 3 + kw] * ib[(size_t)yy * WH_ + xx];
        }
    }
    out[i] = s;
}

// ---------------------------------------------------------------------------
// Row L2 norms of q (y=0) and k (y=1): norms[y][b*64+c] = max(sqrt(sumsq),1e-12)
// ---------------------------------------------------------------------------
__global__ void rownorm(const float* __restrict__ qkv, float* __restrict__ norms) {
    __shared__ float red[256];
    int bc = blockIdx.x;                 // b*64 + c
    int b = bc >> 6, c = bc & 63;
    const float* p = qkv + ((size_t)b * 192 + blockIdx.y * 64 + c) * NPIX;
    float s = 0.f;
    for (int n = threadIdx.x; n < NPIX; n += 256) { float v = p[n]; s += v * v; }
    red[threadIdx.x] = s;
    __syncthreads();
    for (int st = 128; st > 0; st >>= 1) {
        if (threadIdx.x < st) red[threadIdx.x] += red[threadIdx.x + st];
        __syncthreads();
    }
    if (threadIdx.x == 0)
        norms[blockIdx.y * 256 + bc] = fmaxf(sqrtf(red[0]), 1e-12f);
}

// ---------------------------------------------------------------------------
// Gram 8x8 per (b,h) over N=65536, normalize, *temperature, softmax(d).
// One block per (b,h); 64 f32 accumulators per lane, LDS atomic combine.
// ---------------------------------------------------------------------------
__global__ void attn_gram(const float* __restrict__ qkv, const float* __restrict__ norms,
                          const float* __restrict__ temp, float* __restrict__ attn) {
    __shared__ float red[64];
    int bh = blockIdx.x;                 // b*8 + h
    int b = bh >> 3, h = bh & 7;
    const float* q = qkv + ((size_t)b * 192 + h * 8) * NPIX;
    const float* k = qkv + ((size_t)b * 192 + 64 + h * 8) * NPIX;
    float acc[64];
#pragma unroll
    for (int i = 0; i < 64; ++i) acc[i] = 0.f;
    for (int n = threadIdx.x; n < NPIX; n += 256) {
        float qv[8], kv[8];
#pragma unroll
        for (int d = 0; d < 8; ++d) {
            qv[d] = q[(size_t)d * NPIX + n];
            kv[d] = k[(size_t)d * NPIX + n];
        }
#pragma unroll
        for (int c = 0; c < 8; ++c)
#pragma unroll
            for (int d = 0; d < 8; ++d)
                acc[c * 8 + d] += qv[c] * kv[d];
    }
    if (threadIdx.x < 64) red[threadIdx.x] = 0.f;
    __syncthreads();
#pragma unroll
    for (int i = 0; i < 64; ++i) atomicAdd(&red[i], acc[i]);
    __syncthreads();
    if (threadIdx.x < 8) {
        int c = threadIdx.x;
        float t  = temp[h];
        float nq = norms[b * 64 + h * 8 + c];
        float row[8];
        float m = -3.402823466e38f;
#pragma unroll
        for (int d = 0; d < 8; ++d) {
            float nk = norms[256 + b * 64 + h * 8 + d];
            row[d] = red[c * 8 + d] / (nq * nk) * t;
            m = fmaxf(m, row[d]);
        }
        float ssum = 0.f;
#pragma unroll
        for (int d = 0; d < 8; ++d) { row[d] = __expf(row[d] - m); ssum += row[d]; }
        float* ap = attn + (size_t)bh * 64 + c * 8;
#pragma unroll
        for (int d = 0; d < 8; ++d) ap[d] = row[d] / ssum;
    }
}

// ---------------------------------------------------------------------------
// out(b,h*8+c,n) = sum_d attn[b,h,c,d] * v(b,h*8+d,n)*(1+filt(b,h*8+d,n))
// ---------------------------------------------------------------------------
__global__ void attn_apply(const float* __restrict__ qkv, const float* __restrict__ filt,
                           const float* __restrict__ attn, float* __restrict__ outat) {
    __shared__ float sA[64];
    int b = blockIdx.z, h = blockIdx.y;
    int n = blockIdx.x * 256 + threadIdx.x;
    if (threadIdx.x < 64)
        sA[threadIdx.x] = attn[(size_t)(b * 8 + h) * 64 + threadIdx.x];
    __syncthreads();
    const float* v = qkv  + ((size_t)b * 192 + 128 + h * 8) * NPIX + n;
    const float* f = filt + ((size_t)b * 64 + h * 8) * NPIX + n;
    float vm[8];
#pragma unroll
    for (int d = 0; d < 8; ++d)
        vm[d] = v[(size_t)d * NPIX] * (1.f + f[(size_t)d * NPIX]);
    float* o = outat + ((size_t)b * 64 + h * 8) * NPIX + n;
#pragma unroll
    for (int c = 0; c < 8; ++c) {
        float s = 0.f;
#pragma unroll
        for (int d = 0; d < 8; ++d) s += sA[c * 8 + d] * vm[d];
        o[(size_t)c * NPIX] = s;
    }
}

// ---------------------------------------------------------------------------
// Inverse Haar DWT -> out (B,64,512,512). float2 stores.
// ---------------------------------------------------------------------------
__global__ void idwt_haar(const float* __restrict__ LL, const float* __restrict__ LH,
                          const float* __restrict__ HL, const float* __restrict__ HH,
                          float* __restrict__ out) {
    size_t i = (size_t)blockIdx.x * 256 + threadIdx.x;   // over B*64*NPIX
    int xj = (int)(i % WH_);
    int yi = (int)((i / WH_) % HH_);
    size_t bc = i / NPIX;
    float ll = LL[i], lh = LH[i], hl = HL[i], hh = HH[i];
    float a = (ll - lh - hl + hh) * 0.5f;
    float b = (ll - lh + hl - hh) * 0.5f;
    float c = (ll + lh - hl - hh) * 0.5f;
    float d = (ll + lh + hl + hh) * 0.5f;
    float* ob = out + bc * (size_t)HF * WF;
    ((float2*)(ob + (size_t)(2 * yi) * WF))[xj]     = make_float2(a, b);
    ((float2*)(ob + (size_t)(2 * yi + 1) * WF))[xj] = make_float2(c, d);
}

// ---------------------------------------------------------------------------
extern "C" void kernel_launch(void* const* d_in, const int* in_sizes, int n_in,
                              void* d_out, int out_size, void* d_ws, size_t ws_size,
                              hipStream_t stream) {
    (void)in_sizes; (void)n_in; (void)out_size; (void)ws_size;
    const float* x      = (const float*)d_in[0];
    const float* w_hc1  = (const float*)d_in[1];
    const float* w_hc2  = (const float*)d_in[2];
    const float* w_ho   = (const float*)d_in[3];
    const float* w_qkv  = (const float*)d_in[4];
    const float* w_dw   = (const float*)d_in[5];
    const float* w_proj = (const float*)d_in[6];
    const float* temp   = (const float*)d_in[7];
    float* out = (float*)d_out;

    const size_t S = (size_t)BZ * DIMC * NPIX;   // 16,777,216 floats
    float* ws      = (float*)d_ws;
    float* LL      = ws;
    float* LH      = ws + S;
    float* HL      = ws + 2 * S;
    float* HH      = ws + 3 * S;
    float* t1      = ws + 4 * S;    // 2S (hv conv1 out, 128ch)
    float* filt    = ws + 6 * S;    // 1S
    float* qkv_pre = ws + 7 * S;    // 3S
    float* qkv     = ws + 10 * S;   // 3S
    float* outat   = ws + 13 * S;   // 1S
    float* outp    = ws + 14 * S;   // 1S
    float* yh      = ws + 15 * S;   // 3S
    float* norms   = ws + 18 * S;          // 512
    float* attn    = ws + 18 * S + 512;    // 2048

    dwt_haar<<<(int)(S / 256), 256, 0, stream>>>(x, LL, LH, HL, HH);

    // filter_hv = relu(conv1x1(relu(conv3x3_g2([LH,HL]))))
    conv3x3_wmma<<<dim3(WH_ / 64, HH_, BZ * 2), 256, 0, stream>>>(LH, HL, HL, w_hc1, t1, 2, 1);
    conv1x1_wmma<<<dim3(NPIX / 64, 1, BZ), 256, 0, stream>>>(t1, w_hc2, filt, 128, 64, 1);

    // qkv = dwconv3x3(conv1x1(LL))
    conv1x1_wmma<<<dim3(NPIX / 64, 3, BZ), 256, 0, stream>>>(LL, w_qkv, qkv_pre, 64, 192, 0);
    dwconv3x3<<<(int)(((size_t)BZ * 192 * NPIX) / 256), 256, 0, stream>>>(qkv_pre, w_dw, qkv);

    // attention (8x8 per head): norms -> gram+softmax -> apply with v*(1+filt)
    rownorm<<<dim3(256, 2), 256, 0, stream>>>(qkv, norms);
    attn_gram<<<BZ * HEADS, 256, 0, stream>>>(qkv, norms, temp, attn);
    attn_apply<<<dim3(NPIX / 256, HEADS, BZ), 256, 0, stream>>>(qkv, filt, attn, outat);

    // out = conv1x1(out, w_proj)
    conv1x1_wmma<<<dim3(NPIX / 64, 1, BZ), 256, 0, stream>>>(outat, w_proj, outp, 64, 64, 0);

    // yh = relu(conv3x3_g3([LH,HL,HH]))
    conv3x3_wmma<<<dim3(WH_ / 64, HH_, BZ * 3), 256, 0, stream>>>(LH, HL, HH, w_ho, yh, 3, 1);

    // idwt(outp, LH2, HL2, HH2)
    idwt_haar<<<(int)(S / 256), 256, 0, stream>>>(outp, yh, yh + S, yh + 2 * S, out);
}
